// CausalSelfAttention_25572235280979
// MI455X (gfx1250) — matrix operand order repaired and checked by reading, not compiled, not run on hardware
//
#include <hip/hip_runtime.h>

#define DEV __device__ __forceinline__

typedef __attribute__((ext_vector_type(16))) __bf16 v16bf;
typedef __attribute__((ext_vector_type(8)))  float  v8f;

// float -> bf16, round-to-nearest-even
DEV __bf16 f2bf(float f) {
  unsigned u = __builtin_bit_cast(unsigned, f);
  u += 0x7FFFu + ((u >> 16) & 1u);
  unsigned short h = (unsigned short)(u >> 16);
  return __builtin_bit_cast(__bf16, h);
}

DEV v8f wmma_bf16(v16bf a, v16bf b, v8f c) {
#if defined(__gfx1250__)
  // D(16x16,f32) = A(16x32,bf16) x B(32x16,bf16) + C
  return __builtin_amdgcn_wmma_f32_16x16x32_bf16(false, a, false, b, (short)0, c,
                                                 false, false);
#else
  (void)a; (void)b;
  return c;
#endif
}

// A-fragment loader: row-major [16 x 32] bf16 tile with leading dim `ld`.
// Per ISA (16-bit A 16x32): lane L -> row L%16; half=L/16;
// elems 0..7  = K in [8*half, 8*half+8), elems 8..15 = K in [16+8*half, 16+8*half+8).
DEV v16bf load_frag_a(const __bf16* tile, int ld) {
  int lane = threadIdx.x & 31;
  int row = lane & 15, half = lane >> 4;
  const __bf16* p = tile + (size_t)row * ld + half * 8;
  union { v16bf v; uint4 q[2]; } u;
  u.q[0] = *reinterpret_cast<const uint4*>(p);
  u.q[1] = *reinterpret_cast<const uint4*>(p + 16);
  return u.v;
}

// B-fragment loader: B is 32(K) x 16(N); we load from row-major B^T [16 n][32 k].
// Same K order as A: elems 0..7 = K in [8*half, 8*half+8),
// elems 8..15 = K in [16+8*half, 16+8*half+8).
DEV v16bf load_frag_b(const __bf16* tile, int ld) {
  int lane = threadIdx.x & 31;
  int col = lane & 15, half = lane >> 4;
  const __bf16* p = tile + (size_t)col * ld + half * 8;
  union { v16bf v; uint4 q[2]; } u;
  u.q[0] = *reinterpret_cast<const uint4*>(p);
  u.q[1] = *reinterpret_cast<const uint4*>(p + 16);
  return u.v;
}

// ---------------------------------------------------------------- conversions
__global__ void cvt_f32_bf16(const float* __restrict__ in, __bf16* __restrict__ out,
                             int nElem) {
  int i = blockIdx.x * blockDim.x + threadIdx.x;
  if (i < nElem) out[i] = f2bf(in[i]);
}

// out[c][r] = (bf16) in[r][c];  R,C multiples of 32. Block (32,8), grid (C/32, R/32).
__global__ void transpose_cvt(const float* __restrict__ in, __bf16* __restrict__ out,
                              int R, int C) {
  __shared__ float tile[32][33];
  int c0 = blockIdx.x * 32, r0 = blockIdx.y * 32;
  int tx = threadIdx.x, ty = threadIdx.y;
#pragma unroll
  for (int i = 0; i < 32; i += 8)
    tile[ty + i][tx] = in[(size_t)(r0 + ty + i) * C + (c0 + tx)];
  __syncthreads();
#pragma unroll
  for (int i = 0; i < 32; i += 8)
    out[(size_t)(c0 + ty + i) * R + (r0 + tx)] = f2bf(tile[tx][ty + i]);
}

// ------------------------------------------------- fused QKV GEMM + norm/rope
// Xb: [4096][1024] bf16, Wt: [3072][1024] bf16 (W_attn^T), bias fp32 [3072].
// Wave tile: 32 rows x 64 cols; one 64-col tile == one (s,head) -> fuse epilogue.
// Outputs: Qo,Ko bf16 [B*H][T][64]; Vt bf16 [B*H][64][T] (pre-transposed for PV).
__global__ void __launch_bounds__(128)
qkv_gemm_fused(const __bf16* __restrict__ Xb, const __bf16* __restrict__ Wt,
               const float* __restrict__ bias,
               const float* __restrict__ cosT, const float* __restrict__ sinT,
               __bf16* __restrict__ Qo, __bf16* __restrict__ Ko,
               __bf16* __restrict__ Vt) {
  constexpr int Tt = 2048, K = 1024;
  int wid = blockIdx.x * 4 + (threadIdx.x >> 5);
  int mt = wid & 127;        // 128 M-tiles of 32 rows
  int g  = wid >> 7;         // 48 N-tiles of 64 = (s,head)
  int lane = threadIdx.x & 31, half = lane >> 4, n = lane & 15;

  const __bf16* Arow = Xb + (size_t)(mt * 32) * K;
  const __bf16* Brow = Wt + (size_t)(g * 64) * K;

  v8f acc[2][4] = {};
  for (int k0 = 0; k0 < K; k0 += 32) {
    __builtin_prefetch(Arow + k0 + 256, 0, 1);
    __builtin_prefetch(Brow + k0 + 256, 0, 1);
    v16bf a0 = load_frag_a(Arow + k0, K);
    v16bf a1 = load_frag_a(Arow + 16 * K + k0, K);
#pragma unroll
    for (int j = 0; j < 4; ++j) {
      v16bf b = load_frag_b(Brow + (size_t)(j * 16) * K + k0, K);
      acc[0][j] = wmma_bf16(a0, b, acc[0][j]);
      acc[1][j] = wmma_bf16(a1, b, acc[1][j]);
    }
  }

  int s = g >> 4, h = g & 15;
#pragma unroll
  for (int j = 0; j < 4; ++j) {
    float bv = bias[g * 64 + j * 16 + n];
#pragma unroll
    for (int i = 0; i < 2; ++i)
#pragma unroll
      for (int r = 0; r < 8; ++r) acc[i][j][r] += bv;
  }

  if (s == 2) {  // V: plain bias, store transposed [d][t]
#pragma unroll
    for (int i = 0; i < 2; ++i)
#pragma unroll
      for (int r = 0; r < 8; ++r) {
        int m_abs = mt * 32 + i * 16 + r + 8 * half;
        int b = m_abs >> 11, t = m_abs & (Tt - 1);
#pragma unroll
        for (int j = 0; j < 4; ++j) {
          int d = j * 16 + n;
          Vt[(size_t)((b * 16 + h) * 64 + d) * Tt + t] = f2bf(acc[i][j][r]);
        }
      }
    return;
  }

  __bf16* Out = (s == 0) ? Qo : Ko;
#pragma unroll
  for (int i = 0; i < 2; ++i) {
    // RMS norm over the 64-wide head: reduce over j in-lane + 16-lane half.
    float scl[8];
#pragma unroll
    for (int r = 0; r < 8; ++r) {
      float ss = 0.f;
#pragma unroll
      for (int j = 0; j < 4; ++j) ss += acc[i][j][r] * acc[i][j][r];
      ss += __shfl_xor(ss, 1, 32);
      ss += __shfl_xor(ss, 2, 32);
      ss += __shfl_xor(ss, 4, 32);
      ss += __shfl_xor(ss, 8, 32);
      scl[r] = rsqrtf(ss * (1.0f / 64.0f) + 1.1920929e-07f);
    }
#pragma unroll
    for (int j = 0; j < 4; ++j)
#pragma unroll
      for (int r = 0; r < 8; ++r) acc[i][j][r] *= scl[r];

    // Rotary: partner dim d±32 is fragment j±2, same lane/row.
#pragma unroll
    for (int r = 0; r < 8; ++r) {
      int m_abs = mt * 32 + i * 16 + r + 8 * half;
      int b = m_abs >> 11, t = m_abs & (Tt - 1);
      size_t orow = (size_t)((b * 16 + h) * Tt + t) * 64;
#pragma unroll
      for (int j = 0; j < 4; ++j) {
        int d = j * 16 + n;
        float c = cosT[t * 64 + d];
        float sv = sinT[t * 64 + d];
        float rot = (j < 2) ? -acc[i][j + 2][r] : acc[i][j - 2][r];
        Out[orow + d] = f2bf(acc[i][j][r] * c + rot * sv);
      }
    }
  }
}

// ----------------------------------------------------------- flash attention
// One wave per (b,h, 16-row q tile). Q/K bf16 [bh][T][64], Vt bf16 [bh][64][T].
// Y bf16 [B][T][H*64] (row-major A operand for the projection GEMM).
__global__ void __launch_bounds__(32)
flash_attn(const __bf16* __restrict__ Q, const __bf16* __restrict__ Kh,
           const __bf16* __restrict__ Vt, __bf16* __restrict__ Y) {
  constexpr int T = 2048, D = 64;
  __shared__ __align__(16) __bf16 ptile[16 * 40];  // P bounce tile, stride 40
  int wid = blockIdx.x;
  int qt = wid & 127;
  int bh = wid >> 7;
  int lane = threadIdx.x & 31, half = lane >> 4, n = lane & 15;

  const __bf16* qbase = Q + ((size_t)bh * T + qt * 16) * D;
  const __bf16* kbase = Kh + (size_t)bh * T * D;
  const __bf16* vbase = Vt + (size_t)bh * D * T;

  v16bf q0 = load_frag_a(qbase, D);        // d 0..31
  v16bf q1 = load_frag_a(qbase + 32, D);   // d 32..63

  v8f O[4] = {};
  float mi[8], li[8];
#pragma unroll
  for (int r = 0; r < 8; ++r) { mi[r] = -3.0e38f; li[r] = 0.f; }

  int row_base = qt * 16;
  int kb_last = row_base & ~31;  // exactly one masked 32-key tile
  for (int kb = 0; kb <= kb_last; kb += 32) {
    // S = Q K^T  (two 16x16 n-tiles, K=64 over d)
    v8f sf[2];
#pragma unroll
    for (int nt = 0; nt < 2; ++nt) {
      v16bf k0 = load_frag_b(kbase + (size_t)(kb + nt * 16) * D, D);
      v16bf k1 = load_frag_b(kbase + (size_t)(kb + nt * 16) * D + 32, D);
      v8f z = {};
      z = wmma_bf16(q0, k0, z);
      z = wmma_bf16(q1, k1, z);
      sf[nt] = z;
    }

    bool maskt = (kb == kb_last);
#pragma unroll
    for (int nt = 0; nt < 2; ++nt)
#pragma unroll
      for (int r = 0; r < 8; ++r) {
        float v = sf[nt][r] * 0.125f;  // 1/sqrt(64)
        if (maskt) {
          int key = kb + nt * 16 + n;
          int row = row_base + r + 8 * half;
          if (key > row) v = -3.0e38f;
        }
        sf[nt][r] = v;
      }

    // Online softmax: rows r+8*half reduce across the owning 16-lane half.
    float corr[8];
#pragma unroll
    for (int r = 0; r < 8; ++r) {
      float tm = fmaxf(sf[0][r], sf[1][r]);
      tm = fmaxf(tm, __shfl_xor(tm, 1, 32));
      tm = fmaxf(tm, __shfl_xor(tm, 2, 32));
      tm = fmaxf(tm, __shfl_xor(tm, 4, 32));
      tm = fmaxf(tm, __shfl_xor(tm, 8, 32));
      float mn = fmaxf(mi[r], tm);
      corr[r] = __expf(mi[r] - mn);
      mi[r] = mn;
      float p0 = __expf(sf[0][r] - mn);
      float p1 = __expf(sf[1][r] - mn);
      sf[0][r] = p0; sf[1][r] = p1;
      float rs = p0 + p1;
      rs += __shfl_xor(rs, 1, 32);
      rs += __shfl_xor(rs, 2, 32);
      rs += __shfl_xor(rs, 4, 32);
      rs += __shfl_xor(rs, 8, 32);
      li[r] = li[r] * corr[r] + rs;
    }
#pragma unroll
    for (int dt = 0; dt < 4; ++dt)
#pragma unroll
      for (int r = 0; r < 8; ++r) O[dt][r] *= corr[r];

    // C-layout -> A-layout via LDS (single-wave block: barrier is a HW NOP,
    // but fences the compiler and orders the DS ops).
    __syncthreads();
#pragma unroll
    for (int r = 0; r < 8; ++r) {
      int m = r + 8 * half;
      ptile[m * 40 + n]      = f2bf(sf[0][r]);
      ptile[m * 40 + 16 + n] = f2bf(sf[1][r]);
    }
    __syncthreads();
    v16bf pf = load_frag_a(ptile, 40);

    // O += P V   (Vt row-major [d][t] is exactly the B^T layout we need)
#pragma unroll
    for (int dt = 0; dt < 4; ++dt) {
      v16bf vf = load_frag_b(vbase + (size_t)(dt * 16) * T + kb, T);
      O[dt] = wmma_bf16(pf, vf, O[dt]);
    }
  }

  int b = bh >> 4, h = bh & 15;
#pragma unroll
  for (int r = 0; r < 8; ++r) {
    int m = r + 8 * half;
    int t = row_base + m;
    float inv = 1.0f / li[r];
    size_t yrow = (size_t)(b * T + t) * 1024 + h * 64;
#pragma unroll
    for (int dt = 0; dt < 4; ++dt)
      Y[yrow + dt * 16 + n] = f2bf(O[dt][r] * inv);
  }
}

// -------------------------------------------------- projection GEMM (+ bias)
__global__ void __launch_bounds__(128)
gemm_bias_f32(const __bf16* __restrict__ A, const __bf16* __restrict__ Bt,
              const float* __restrict__ bias, float* __restrict__ Cout,
              int M, int N, int K) {
  int wid = blockIdx.x * 4 + (threadIdx.x >> 5);
  int MT = M / 32;
  int mt = wid % MT, nt = wid / MT;
  int lane = threadIdx.x & 31, half = lane >> 4, n = lane & 15;
  const __bf16* Arow = A + (size_t)(mt * 32) * K;
  const __bf16* Brow = Bt + (size_t)(nt * 64) * K;
  v8f acc[2][4] = {};
  for (int k0 = 0; k0 < K; k0 += 32) {
    __builtin_prefetch(Arow + k0 + 256, 0, 1);
    __builtin_prefetch(Brow + k0 + 256, 0, 1);
    v16bf a0 = load_frag_a(Arow + k0, K);
    v16bf a1 = load_frag_a(Arow + 16 * K + k0, K);
#pragma unroll
    for (int j = 0; j < 4; ++j) {
      v16bf b = load_frag_b(Brow + (size_t)(j * 16) * K + k0, K);
      acc[0][j] = wmma_bf16(a0, b, acc[0][j]);
      acc[1][j] = wmma_bf16(a1, b, acc[1][j]);
    }
  }
#pragma unroll
  for (int j = 0; j < 4; ++j) {
    int gn = nt * 64 + j * 16 + n;
    float bv = bias[gn];
#pragma unroll
    for (int i = 0; i < 2; ++i)
#pragma unroll
      for (int r = 0; r < 8; ++r) {
        int gm = mt * 32 + i * 16 + r + 8 * half;
        Cout[(size_t)gm * N + gn] = acc[i][j][r] + bv;
      }
  }
}

// ---------------------------------------------------------------- launch glue
extern "C" void kernel_launch(void* const* d_in, const int* in_sizes, int n_in,
                              void* d_out, int out_size, void* d_ws, size_t ws_size,
                              hipStream_t stream) {
  (void)in_sizes; (void)n_in; (void)out_size; (void)ws_size;
  const float* x      = (const float*)d_in[0];
  const float* cosT   = (const float*)d_in[1];
  const float* sinT   = (const float*)d_in[2];
  const float* W_attn = (const float*)d_in[3];
  const float* b_attn = (const float*)d_in[4];
  const float* W_proj = (const float*)d_in[5];
  const float* b_proj = (const float*)d_in[6];
  float* out = (float*)d_out;

  char* ws = (char*)d_ws;
  const size_t MB = (size_t)1 << 20;
  __bf16* xb  = (__bf16*)(ws);             //  8 MB  x bf16 [4096][1024]
  __bf16* wat = (__bf16*)(ws + 8 * MB);    //  6 MB  W_attn^T bf16 [3072][1024]
  __bf16* wpt = (__bf16*)(ws + 14 * MB);   //  2 MB  W_proj^T bf16 [1024][1024]
  __bf16* qb  = (__bf16*)(ws + 16 * MB);   //  8 MB  q bf16 [32][2048][64]
  __bf16* kb  = (__bf16*)(ws + 24 * MB);   //  8 MB  k bf16 [32][2048][64]
  __bf16* vt  = (__bf16*)(ws + 32 * MB);   //  8 MB  v^T bf16 [32][64][2048]
  __bf16* yb  = (__bf16*)(ws + 40 * MB);   //  8 MB  attn out bf16 [4096][1024]

  cvt_f32_bf16<<<(4096 * 1024) / 256, 256, 0, stream>>>(x, xb, 4096 * 1024);
  transpose_cvt<<<dim3(3072 / 32, 1024 / 32), dim3(32, 8), 0, stream>>>(
      W_attn, wat, 1024, 3072);
  transpose_cvt<<<dim3(1024 / 32, 1024 / 32), dim3(32, 8), 0, stream>>>(
      W_proj, wpt, 1024, 1024);
  qkv_gemm_fused<<<(128 * 48) / 4, 128, 0, stream>>>(xb, wat, b_attn, cosT, sinT,
                                                     qb, kb, vt);
  flash_attn<<<32 * 128, 32, 0, stream>>>(qb, kb, vt, yb);
  gemm_bias_f32<<<(128 * 16) / 4, 128, 0, stream>>>(yb, wpt, b_proj, out,
                                                    4096, 1024, 1024);
}
